// VectorFieldEstimator_69784628626079
// MI455X (gfx1250) — compile-verified, matrix operand order checked by reading, not run on hardware
//
#include <hip/hip_runtime.h>

// ---------------- problem constants ----------------
#define BB 8
#define TT 2048
#define LL 512
#define CC 512
#define HH 8
#define HDs 64
#define EE 1024
#define NB 3
#define EPSf 1e-6f

typedef unsigned short u16;
typedef unsigned int u32;
typedef __attribute__((ext_vector_type(16))) __bf16 v16bf;
typedef __attribute__((ext_vector_type(8)))  float  v8f;

__device__ __forceinline__ u16 f2bf(float f) {
  u32 u = __float_as_uint(f);
  u32 r = (u + 0x7FFFu + ((u >> 16) & 1u)) >> 16;   // round-to-nearest-even
  return (u16)r;
}

union Frag { uint4 q[2]; v16bf v; };

// =====================================================================
// WMMA GEMM, 128x128 block tile, 256 threads (8 waves), K-step 32.
// C = alpha * A(MxK) * B(NxK)^T + bias ; A,B bf16 row-major; batched over z.
// Requires M % 128 == 0, N % 128 == 0, K % 32 == 0.
// Wave w (0..7): rows [ (w>>2)*64 , +64 ), cols [ (w&3)*32 , +32 )
//  -> 4x2 = 8 v_wmma per k-step, 6 fragment loads (4 A + 2 B).
// =====================================================================
template<bool OUT_BF16, bool DO_GELU>
__global__ __launch_bounds__(256) void gemm_nt_128(
    const u16* __restrict__ A, const u16* __restrict__ B, void* __restrict__ C,
    const float* __restrict__ bias, int M, int N, int Kd,
    int lda, int ldb, int ldc,
    long long sAz, long long sBz, long long sCz, float alpha)
{
  __shared__ __align__(16) u16 As[128][40];
  __shared__ __align__(16) u16 Bs[128][40];
  const int tid  = threadIdx.x;
  const int lane = tid & 31;
  const int wave = tid >> 5;
  const long long z = blockIdx.z;
  const u16* Ab = A + z * sAz;
  const u16* Bb = B + z * sBz;
  const int m0 = blockIdx.y * 128;
  const int n0 = blockIdx.x * 128;
  const int wm = (wave >> 2) * 64;
  const int wn = (wave & 3) * 32;

  v8f acc[4][2];
#pragma unroll
  for (int s = 0; s < 4; s++)
#pragma unroll
    for (int t = 0; t < 2; t++)
      acc[s][t] = (v8f){0.f,0.f,0.f,0.f,0.f,0.f,0.f,0.f};

  const int lrow = tid >> 1;          // 0..127
  const int lcol = (tid & 1) << 4;    // 0 or 16 (halves)

  for (int k0 = 0; k0 < Kd; k0 += 32) {
    const u16* ga = Ab + (long long)(m0 + lrow) * lda + (k0 + lcol);
    const u16* gb = Bb + (long long)(n0 + lrow) * ldb + (k0 + lcol);
    uint4 a0 = *(const uint4*)ga;
    uint4 a1 = *(const uint4*)(ga + 8);
    uint4 b0 = *(const uint4*)gb;
    uint4 b1 = *(const uint4*)(gb + 8);
    __syncthreads();
    *(uint4*)&As[lrow][lcol]     = a0;
    *(uint4*)&As[lrow][lcol + 8] = a1;
    *(uint4*)&Bs[lrow][lcol]     = b0;
    *(uint4*)&Bs[lrow][lcol + 8] = b1;
    __syncthreads();

    const int kb = (lane >> 4) << 3;  // 0 or 8 (per-lane K sub-block)
    Frag fa[4], fb[2];
#pragma unroll
    for (int s = 0; s < 4; s++) {
      int r = wm + s * 16 + (lane & 15);
      fa[s].q[0] = *(const uint4*)&As[r][kb];
      fa[s].q[1] = *(const uint4*)&As[r][kb + 16];
    }
#pragma unroll
    for (int t = 0; t < 2; t++) {
      int r = wn + t * 16 + (lane & 15);
      fb[t].q[0] = *(const uint4*)&Bs[r][kb];
      fb[t].q[1] = *(const uint4*)&Bs[r][kb + 16];
    }
#pragma unroll
    for (int s = 0; s < 4; s++)
#pragma unroll
      for (int t = 0; t < 2; t++)
        acc[s][t] = __builtin_amdgcn_wmma_f32_16x16x32_bf16(
            false, fa[s].v, false, fb[t].v, (short)0, acc[s][t], false, false);
  }

  const int mo    = (lane >> 4) << 3;  // +8 rows for lanes 16..31
  const int nlane = lane & 15;
#pragma unroll
  for (int s = 0; s < 4; s++) {
#pragma unroll
    for (int t = 0; t < 2; t++) {
#pragma unroll
      for (int j = 0; j < 8; j++) {
        int m = m0 + wm + s * 16 + mo + j;
        int n = n0 + wn + t * 16 + nlane;
        float v = acc[s][t][j] * alpha;
        if (bias) v += bias[n];
        if (DO_GELU) v = 0.5f * v * (1.0f + erff(v * 0.70710678118654752f));
        long long offo = z * sCz + (long long)m * ldc + n;
        if (OUT_BF16) ((u16*)C)[offo] = f2bf(v);
        else          ((float*)C)[offo] = v;
      }
    }
  }
}

// =====================================================================
// WMMA GEMM, 64x64 block tile, 128 threads (4 waves) — for N == 64 case
// (attn @ V). Same semantics as above; M%64==0, N==64 (grid.x==1), K%32==0.
// =====================================================================
template<bool OUT_BF16, bool DO_GELU>
__global__ __launch_bounds__(128) void gemm_nt_64(
    const u16* __restrict__ A, const u16* __restrict__ B, void* __restrict__ C,
    const float* __restrict__ bias, int M, int N, int Kd,
    int lda, int ldb, int ldc,
    long long sAz, long long sBz, long long sCz, float alpha)
{
  __shared__ __align__(16) u16 As[64][40];
  __shared__ __align__(16) u16 Bs[64][40];
  const int tid  = threadIdx.x;
  const int lane = tid & 31;
  const int wave = tid >> 5;
  const long long z = blockIdx.z;
  const u16* Ab = A + z * sAz;
  const u16* Bb = B + z * sBz;
  const int m0 = blockIdx.y * 64;
  const int n0 = blockIdx.x * 64;
  const int wm = (wave >> 1) * 32;
  const int wn = (wave & 1) * 32;

  v8f acc[2][2];
#pragma unroll
  for (int s = 0; s < 2; s++)
#pragma unroll
    for (int t = 0; t < 2; t++)
      acc[s][t] = (v8f){0.f,0.f,0.f,0.f,0.f,0.f,0.f,0.f};

  const int lrow = tid >> 1;
  const int lcol = (tid & 1) << 4;

  for (int k0 = 0; k0 < Kd; k0 += 32) {
    const u16* ga = Ab + (long long)(m0 + lrow) * lda + (k0 + lcol);
    const u16* gb = Bb + (long long)(n0 + lrow) * ldb + (k0 + lcol);
    uint4 a0 = *(const uint4*)ga;
    uint4 a1 = *(const uint4*)(ga + 8);
    uint4 b0 = *(const uint4*)gb;
    uint4 b1 = *(const uint4*)(gb + 8);
    __syncthreads();
    *(uint4*)&As[lrow][lcol]     = a0;
    *(uint4*)&As[lrow][lcol + 8] = a1;
    *(uint4*)&Bs[lrow][lcol]     = b0;
    *(uint4*)&Bs[lrow][lcol + 8] = b1;
    __syncthreads();

    const int kb = (lane >> 4) << 3;
    Frag fa[2], fb[2];
#pragma unroll
    for (int s = 0; s < 2; s++) {
      int r = wm + s * 16 + (lane & 15);
      fa[s].q[0] = *(const uint4*)&As[r][kb];
      fa[s].q[1] = *(const uint4*)&As[r][kb + 16];
    }
#pragma unroll
    for (int t = 0; t < 2; t++) {
      int r = wn + t * 16 + (lane & 15);
      fb[t].q[0] = *(const uint4*)&Bs[r][kb];
      fb[t].q[1] = *(const uint4*)&Bs[r][kb + 16];
    }
#pragma unroll
    for (int s = 0; s < 2; s++)
#pragma unroll
      for (int t = 0; t < 2; t++)
        acc[s][t] = __builtin_amdgcn_wmma_f32_16x16x32_bf16(
            false, fa[s].v, false, fb[t].v, (short)0, acc[s][t], false, false);
  }

  const int mo    = (lane >> 4) << 3;
  const int nlane = lane & 15;
#pragma unroll
  for (int s = 0; s < 2; s++) {
#pragma unroll
    for (int t = 0; t < 2; t++) {
#pragma unroll
      for (int j = 0; j < 8; j++) {
        int m = m0 + wm + s * 16 + mo + j;
        int n = n0 + wn + t * 16 + nlane;
        float v = acc[s][t][j] * alpha;
        if (bias) v += bias[n];
        if (DO_GELU) v = 0.5f * v * (1.0f + erff(v * 0.70710678118654752f));
        long long offo = z * sCz + (long long)m * ldc + n;
        if (OUT_BF16) ((u16*)C)[offo] = f2bf(v);
        else          ((float*)C)[offo] = v;
      }
    }
  }
}

// ---------------- sequence lengths from masks ----------------
__global__ __launch_bounds__(256) void lens_kernel(const float* __restrict__ xm,
                                                   const float* __restrict__ cm,
                                                   float* __restrict__ lens) {
  __shared__ float sb[256];
  int b = blockIdx.x, tid = threadIdx.x;
  float s = 0.f;
  for (int t = tid; t < TT; t += 256) s += xm[b * TT + t];
  sb[tid] = s; __syncthreads();
  for (int k = 128; k > 0; k >>= 1) { if (tid < k) sb[tid] += sb[tid + k]; __syncthreads(); }
  if (tid == 0) lens[b] = sb[0];
  __syncthreads();
  s = 0.f;
  for (int t = tid; t < LL; t += 256) s += cm[b * LL + t];
  sb[tid] = s; __syncthreads();
  for (int k = 128; k > 0; k >>= 1) { if (tid < k) sb[tid] += sb[tid + k]; __syncthreads(); }
  if (tid == 0) lens[BB + b] = sb[0];
}

// ---------------- x (B,C,T) -> masked (B,T,C) f32 + bf16 ----------------
__global__ __launch_bounds__(256) void xpose_in(const float* __restrict__ x,
                                                const float* __restrict__ xm,
                                                float* __restrict__ Xf,
                                                u16* __restrict__ Xb) {
  __shared__ float tile[32][33];
  int b = blockIdx.z;
  int t0 = blockIdx.x * 32, c0 = blockIdx.y * 32;
  int tx = threadIdx.x, ty = threadIdx.y;
#pragma unroll
  for (int i = 0; i < 4; i++)
    tile[ty + i * 8][tx] = x[((long long)b * CC + (c0 + ty + i * 8)) * TT + t0 + tx];
  __syncthreads();
#pragma unroll
  for (int i = 0; i < 4; i++) {
    int t = t0 + ty + i * 8, c = c0 + tx;
    float v = tile[tx][ty + i * 8] * xm[b * TT + t];
    long long o = ((long long)b * TT + t) * CC + c;
    Xf[o] = v; Xb[o] = f2bf(v);
  }
}

// ---------------- final (B,T,C) f32 -> (B,C,T) ----------------
__global__ __launch_bounds__(256) void xpose_out(const float* __restrict__ Xf,
                                                 float* __restrict__ out) {
  __shared__ float tile[32][33];
  int b = blockIdx.z;
  int t0 = blockIdx.x * 32, c0 = blockIdx.y * 32;
  int tx = threadIdx.x, ty = threadIdx.y;
#pragma unroll
  for (int i = 0; i < 4; i++)
    tile[ty + i * 8][tx] = Xf[((long long)b * TT + (t0 + ty + i * 8)) * CC + c0 + tx];
  __syncthreads();
#pragma unroll
  for (int i = 0; i < 4; i++)
    out[((long long)b * CC + (c0 + ty + i * 8)) * TT + t0 + tx] = tile[tx][ty + i * 8];
}

// ---------------- weight transpose (R x Cl) f32 -> (Cl x R) bf16 -----------
__global__ __launch_bounds__(256) void wpose(const float* __restrict__ W,
                                             u16* __restrict__ Wt, int R, int Cl) {
  __shared__ float tile[32][33];
  int r0 = blockIdx.x * 32, c0 = blockIdx.y * 32;
  int tx = threadIdx.x, ty = threadIdx.y;
#pragma unroll
  for (int i = 0; i < 4; i++)
    tile[ty + i * 8][tx] = W[(long long)(r0 + ty + i * 8) * Cl + c0 + tx];
  __syncthreads();
#pragma unroll
  for (int i = 0; i < 4; i++)
    Wt[(long long)(c0 + ty + i * 8) * R + r0 + tx] = f2bf(tile[tx][ty + i * 8]);
}

__global__ void cvt_bf16(const float* __restrict__ in, u16* __restrict__ out, long long n) {
  long long i = (long long)blockIdx.x * 256 + threadIdx.x;
  if (i < n) out[i] = f2bf(in[i]);
}

// ---------------- RoPE: f32 proj -> bf16, per (b, pos) ----------------
__global__ __launch_bounds__(256) void rope_kernel(const float* __restrict__ Qf,
                                                   u16* __restrict__ Qb,
                                                   const float* __restrict__ lens,
                                                   int lenDim, int lensOff) {
  int b = blockIdx.y, t = blockIdx.x, tid = threadIdx.x;
  int h = tid >> 5, j = tid & 31;
  float len = lens[lensOff + b];
  float theta = 10.0f * powf(10000.0f, -(float)(2 * j) / 64.0f);
  float f = (float)t / len * theta;
  float c = cosf(f), s = sinf(f);
  long long base = ((long long)b * lenDim + t) * CC + h * HDs;
  float q1 = Qf[base + j], q2 = Qf[base + 32 + j];
  Qb[base + j]      = f2bf(q1 * c - q2 * s);
  Qb[base + 32 + j] = f2bf(q1 * s + q2 * c);
}

// ---------------- V (B,L,H,HD) f32 -> Vt (B,H,HD,L) bf16 ----------------
__global__ void vtrans(const float* __restrict__ Vf, u16* __restrict__ Vt) {
  long long i = (long long)blockIdx.x * 256 + threadIdx.x;
  if (i >= (long long)BB * HH * HDs * LL) return;
  int l = (int)(i & (LL - 1));
  long long r = i >> 9;
  int d = (int)(r & (HDs - 1));
  long long r2 = r >> 6;
  int h = (int)(r2 & (HH - 1));
  int b = (int)(r2 >> 3);
  Vt[i] = f2bf(Vf[(((long long)b * LL + l) * CC) + h * HDs + d]);
}

// ---------------- masked softmax, logits(H,T,L) f32 -> attn bf16 ----------
__global__ __launch_bounds__(128) void softmax_kernel(const float* __restrict__ logits,
                                                      u16* __restrict__ attn,
                                                      const float* __restrict__ lens, int b) {
  __shared__ float sb[128];
  int t = blockIdx.x, h = blockIdx.y, tid = threadIdx.x;
  int lq = (int)(lens[b] + 0.5f);
  int lk = (int)(lens[BB + b] + 0.5f);
  const float* row  = logits + ((long long)h * TT + t) * LL;
  u16*         orow = attn   + ((long long)h * TT + t) * LL;
  if (t >= lq) {
    for (int i = 0; i < 4; i++) orow[tid + i * 128] = 0;
    return;
  }
  float vals[4]; float mx = -3.0e38f;
#pragma unroll
  for (int i = 0; i < 4; i++) {
    int l = tid + i * 128;
    vals[i] = (l < lk) ? row[l] : -3.0e38f;
    mx = fmaxf(mx, vals[i]);
  }
  sb[tid] = mx; __syncthreads();
  for (int k = 64; k > 0; k >>= 1) { if (tid < k) sb[tid] = fmaxf(sb[tid], sb[tid + k]); __syncthreads(); }
  mx = sb[0]; __syncthreads();
  float ss = 0.f;
#pragma unroll
  for (int i = 0; i < 4; i++) {
    int l = tid + i * 128;
    vals[i] = (l < lk) ? expf(vals[i] - mx) : 0.0f;
    ss += vals[i];
  }
  sb[tid] = ss; __syncthreads();
  for (int k = 64; k > 0; k >>= 1) { if (tid < k) sb[tid] += sb[tid + k]; __syncthreads(); }
  float inv = 1.0f / sb[0];
#pragma unroll
  for (int i = 0; i < 4; i++) orow[tid + i * 128] = f2bf(vals[i] * inv);
}

// ------------- residual add + channel layernorm + mask (in place on Xf) ----
__global__ __launch_bounds__(128) void ln_add_kernel(float* __restrict__ Xf,
                                                     const float* __restrict__ Of,
                                                     const float* __restrict__ g,
                                                     const float* __restrict__ bt,
                                                     const float* __restrict__ lens) {
  __shared__ float sb[128];
  int t = blockIdx.x, b = blockIdx.y, tid = threadIdx.x;
  int lq = (int)(lens[b] + 0.5f);
  long long base = ((long long)b * TT + t) * CC;
  if (t >= lq) {
    for (int i = 0; i < 4; i++) Xf[base + tid + i * 128] = 0.0f;
    return;
  }
  float v[4]; float s = 0.f;
#pragma unroll
  for (int i = 0; i < 4; i++) { v[i] = Xf[base + tid + i * 128] + Of[base + tid + i * 128]; s += v[i]; }
  sb[tid] = s; __syncthreads();
  for (int k = 64; k > 0; k >>= 1) { if (tid < k) sb[tid] += sb[tid + k]; __syncthreads(); }
  float m = sb[0] * (1.0f / CC); __syncthreads();
  float q = 0.f;
#pragma unroll
  for (int i = 0; i < 4; i++) { float d = v[i] - m; q += d * d; }
  sb[tid] = q; __syncthreads();
  for (int k = 64; k > 0; k >>= 1) { if (tid < k) sb[tid] += sb[tid + k]; __syncthreads(); }
  float r = rsqrtf(sb[0] * (1.0f / CC) + EPSf);
#pragma unroll
  for (int i = 0; i < 4; i++) {
    int c = tid + i * 128;
    Xf[base + c] = (v[i] - m) * r * g[c] + bt[c];
  }
}

// ---------- depthwise dilated conv (edge pad) + mask + channel-LN -> bf16 ---
__global__ __launch_bounds__(128) void dwconv_ln_kernel(const float* __restrict__ Xf,
                                                        u16* __restrict__ LNh,
                                                        const float* __restrict__ dww,
                                                        const float* __restrict__ dwb,
                                                        const float* __restrict__ g,
                                                        const float* __restrict__ bt,
                                                        const float* __restrict__ lens,
                                                        int iblk, int dil) {
  __shared__ float sb[128];
  int t = blockIdx.x, b = blockIdx.y, tid = threadIdx.x;
  int lq = (int)(lens[b] + 0.5f);
  float v[4]; float s = 0.f;
#pragma unroll
  for (int i = 0; i < 4; i++) {
    int c = tid + i * 128;
    float acc = 0.0f;
    if (t < lq) {
      acc = dwb[iblk * CC + c];
#pragma unroll
      for (int k = 0; k < 5; k++) {
        int tt = t + (k - 2) * dil;
        tt = tt < 0 ? 0 : (tt > TT - 1 ? TT - 1 : tt);
        acc += dww[((long long)iblk * CC + c) * 5 + k] * Xf[((long long)b * TT + tt) * CC + c];
      }
    }
    v[i] = acc; s += acc;
  }
  sb[tid] = s; __syncthreads();
  for (int k = 64; k > 0; k >>= 1) { if (tid < k) sb[tid] += sb[tid + k]; __syncthreads(); }
  float m = sb[0] * (1.0f / CC); __syncthreads();
  float q = 0.f;
#pragma unroll
  for (int i = 0; i < 4; i++) { float d = v[i] - m; q += d * d; }
  sb[tid] = q; __syncthreads();
  for (int k = 64; k > 0; k >>= 1) { if (tid < k) sb[tid] += sb[tid + k]; __syncthreads(); }
  float r = rsqrtf(sb[0] * (1.0f / CC) + EPSf);
  long long base = ((long long)b * TT + t) * CC;
#pragma unroll
  for (int i = 0; i < 4; i++) {
    int c = tid + i * 128;
    LNh[base + c] = f2bf((v[i] - m) * r * g[iblk * CC + c] + bt[iblk * CC + c]);
  }
}

// ---------------- Xf = mask * (gamma * H2 + Xf) ----------------
__global__ void resid_kernel(float* __restrict__ Xf, const float* __restrict__ H2,
                             const float* __restrict__ gamma,
                             const float* __restrict__ lens, int iblk) {
  long long i = (long long)blockIdx.x * 256 + threadIdx.x;
  if (i >= (long long)BB * TT * CC) return;
  int c = (int)(i & (CC - 1));
  long long rr = i >> 9;
  int t = (int)(rr & (TT - 1));
  int b = (int)(rr >> 11);
  int lq = (int)(lens[b] + 0.5f);
  Xf[i] = (t < lq) ? (gamma[iblk * CC + c] * H2[i] + Xf[i]) : 0.0f;
}

// =======================================================================
extern "C" void kernel_launch(void* const* d_in, const int* in_sizes, int n_in,
                              void* d_out, int out_size, void* d_ws, size_t ws_size,
                              hipStream_t stream) {
  const float* x    = (const float*)d_in[0];
  const float* ctx  = (const float*)d_in[1];
  const float* xm   = (const float*)d_in[2];
  const float* cm   = (const float*)d_in[3];
  const float* Wq   = (const float*)d_in[4];
  const float* bq   = (const float*)d_in[5];
  const float* Wk   = (const float*)d_in[6];
  const float* bk   = (const float*)d_in[7];
  const float* Wv   = (const float*)d_in[8];
  const float* bv   = (const float*)d_in[9];
  const float* Wo   = (const float*)d_in[10];
  const float* bo   = (const float*)d_in[11];
  const float* lng  = (const float*)d_in[12];
  const float* lnb  = (const float*)d_in[13];
  const float* dww  = (const float*)d_in[14];
  const float* dwb  = (const float*)d_in[15];
  const float* cng  = (const float*)d_in[16];
  const float* cnb  = (const float*)d_in[17];
  const float* pw1w = (const float*)d_in[18];
  const float* pw1b = (const float*)d_in[19];
  const float* pw2w = (const float*)d_in[20];
  const float* pw2b = (const float*)d_in[21];
  const float* cgam = (const float*)d_in[22];

  char* ws = (char*)d_ws;
  size_t off = 0;
  auto take = [&](size_t bytes) -> char* {
    off = (off + 255) & ~(size_t)255;
    char* p = ws + off; off += bytes; return p;
  };

  float* F32A = (float*)take(sizeof(float) * (size_t)BB * TT * CC); // Qf / logits / Of / H2f
  float* Xf   = (float*)take(sizeof(float) * (size_t)BB * TT * CC); // residual / running activation
  float* Kf   = (float*)take(sizeof(float) * (size_t)BB * LL * CC);
  float* Vf   = (float*)take(sizeof(float) * (size_t)BB * LL * CC);
  u16*   Xb   = (u16*)take(2ull * BB * TT * CC);                    // x bf16, reused as attn-out bf16
  u16*   Cb   = (u16*)take(2ull * BB * LL * CC);
  u16*   Qb   = (u16*)take(2ull * BB * TT * CC);
  u16*   Kb   = (u16*)take(2ull * BB * LL * CC);
  u16*   Vt   = (u16*)take(2ull * BB * HH * HDs * LL);
  u16*   ATb  = (u16*)take(2ull * HH * TT * LL);                    // attn per-b, reused as LN-h bf16
  u16*   H1b  = (u16*)take(2ull * BB * TT * EE);
  u16*   WqT  = (u16*)take(2ull * CC * CC);
  u16*   WkT  = (u16*)take(2ull * CC * CC);
  u16*   WvT  = (u16*)take(2ull * CC * CC);
  u16*   WoT  = (u16*)take(2ull * CC * CC);
  u16*   P1b  = (u16*)take(2ull * NB * EE * CC);
  u16*   P2b  = (u16*)take(2ull * NB * CC * EE);
  float* lens = (float*)take(sizeof(float) * 2 * BB);

  // ---- prep ----
  lens_kernel<<<BB, 256, 0, stream>>>(xm, cm, lens);
  xpose_in<<<dim3(TT / 32, CC / 32, BB), dim3(32, 8), 0, stream>>>(x, xm, Xf, Xb);
  { long long n = (long long)BB * LL * CC; cvt_bf16<<<(n + 255) / 256, 256, 0, stream>>>(ctx, Cb, n); }
  wpose<<<dim3(CC / 32, CC / 32), dim3(32, 8), 0, stream>>>(Wq, WqT, CC, CC);
  wpose<<<dim3(CC / 32, CC / 32), dim3(32, 8), 0, stream>>>(Wk, WkT, CC, CC);
  wpose<<<dim3(CC / 32, CC / 32), dim3(32, 8), 0, stream>>>(Wv, WvT, CC, CC);
  wpose<<<dim3(CC / 32, CC / 32), dim3(32, 8), 0, stream>>>(Wo, WoT, CC, CC);
  { long long n = (long long)NB * EE * CC; cvt_bf16<<<(n + 255) / 256, 256, 0, stream>>>(pw1w, P1b, n); }
  { long long n = (long long)NB * CC * EE; cvt_bf16<<<(n + 255) / 256, 256, 0, stream>>>(pw2w, P2b, n); }

  // ---- Q/K/V projections (WMMA, 128x128 tiles) ----
  gemm_nt_128<false, false><<<dim3(CC / 128, (BB * TT) / 128, 1), 256, 0, stream>>>(
      Xb, WqT, F32A, bq, BB * TT, CC, CC, CC, CC, CC, 0, 0, 0, 1.0f);
  gemm_nt_128<false, false><<<dim3(CC / 128, (BB * LL) / 128, 1), 256, 0, stream>>>(
      Cb, WkT, Kf, bk, BB * LL, CC, CC, CC, CC, CC, 0, 0, 0, 1.0f);
  gemm_nt_128<false, false><<<dim3(CC / 128, (BB * LL) / 128, 1), 256, 0, stream>>>(
      Cb, WvT, Vf, bv, BB * LL, CC, CC, CC, CC, CC, 0, 0, 0, 1.0f);

  // ---- RoPE + V transpose ----
  rope_kernel<<<dim3(TT, BB), 256, 0, stream>>>(F32A, Qb, lens, TT, 0);
  rope_kernel<<<dim3(LL, BB), 256, 0, stream>>>(Kf, Kb, lens, LL, BB);
  { long long n = (long long)BB * HH * HDs * LL; vtrans<<<(n + 255) / 256, 256, 0, stream>>>(Vf, Vt); }

  // ---- attention per batch (logits buffer reused; batched over heads) ----
  const float inv_scale = 0.04419417382415922f;  // 1/sqrt(512)
  for (int b = 0; b < BB; b++) {
    gemm_nt_128<false, false><<<dim3(LL / 128, TT / 128, HH), 256, 0, stream>>>(
        Qb + (long long)b * TT * CC, Kb + (long long)b * LL * CC, F32A, nullptr,
        TT, LL, HDs, CC, CC, LL,
        (long long)HDs, (long long)HDs, (long long)TT * LL, inv_scale);
    softmax_kernel<<<dim3(TT, HH), 128, 0, stream>>>(F32A, ATb, lens, b);
    gemm_nt_64<true, false><<<dim3(1, TT / 64, HH), 128, 0, stream>>>(
        ATb, Vt + (long long)b * HH * HDs * LL, Xb + (long long)b * TT * CC, nullptr,
        TT, HDs, LL, LL, LL, CC,
        (long long)TT * LL, (long long)HDs * LL, (long long)HDs, 1.0f);
  }

  // ---- output projection + residual + channel-LN ----
  gemm_nt_128<false, false><<<dim3(CC / 128, (BB * TT) / 128, 1), 256, 0, stream>>>(
      Xb, WoT, F32A, bo, BB * TT, CC, CC, CC, CC, CC, 0, 0, 0, 1.0f);
  ln_add_kernel<<<dim3(TT, BB), 128, 0, stream>>>(Xf, F32A, lng, lnb, lens);

  // ---- 3 dilated conv-FFN blocks ----
  const int dils[NB] = {1, 2, 4};
  for (int i = 0; i < NB; i++) {
    dwconv_ln_kernel<<<dim3(TT, BB), 128, 0, stream>>>(Xf, ATb, dww, dwb, cng, cnb, lens, i, dils[i]);
    gemm_nt_128<true, true><<<dim3(EE / 128, (BB * TT) / 128, 1), 256, 0, stream>>>(
        ATb, P1b + (long long)i * EE * CC, H1b, pw1b + i * EE,
        BB * TT, EE, CC, CC, CC, EE, 0, 0, 0, 1.0f);
    gemm_nt_128<false, false><<<dim3(CC / 128, (BB * TT) / 128, 1), 256, 0, stream>>>(
        H1b, P2b + (long long)i * CC * EE, F32A, pw2b + i * CC,
        BB * TT, CC, EE, EE, EE, CC, 0, 0, 0, 1.0f);
    { long long n = (long long)BB * TT * CC;
      resid_kernel<<<(n + 255) / 256, 256, 0, stream>>>(Xf, F32A, cgam, lens, i); }
  }

  // ---- final transpose to (B, C, T) ----
  xpose_out<<<dim3(TT / 32, CC / 32, BB), dim3(32, 8), 0, stream>>>(Xf, (float*)d_out);
}
